// MultiHeadAttention_69861938037649
// MI455X (gfx1250) — matrix operand order repaired and checked by reading, not compiled, not run on hardware
//
#include <hip/hip_runtime.h>

// ---------------------------------------------------------------------------
// MultiHeadAttention fused pipeline for MI455X (gfx1250, wave32, WMMA).
// B=4, S=2048, D=512, H=8, hd=64.  All matmuls via v_wmma_f32_16x16x32_bf16.
// v3: GEMM K-loop ping-pong double buffer (2x unroll, no register rotation),
//     early-issued attention fragment loads + global_prefetch, wide LN passes.
// ---------------------------------------------------------------------------

typedef unsigned short u16;
typedef __attribute__((ext_vector_type(8)))  float  v8f;
typedef __attribute__((ext_vector_type(8)))  __bf16 v8bf;
typedef __attribute__((ext_vector_type(16))) __bf16 v16bf;

#define NB 4
#define NS 2048
#define ND 512
#define NH 8
#define HD 64

static __device__ __forceinline__ u16 f2bf(float f) {
  unsigned int u = __float_as_uint(f);
  u += 0x7FFFu + ((u >> 16) & 1u);          // round-to-nearest-even
  return (u16)(u >> 16);
}

static __device__ __forceinline__ v16bf combine16(v8bf lo, v8bf hi) {
  return __builtin_shufflevector(lo, hi, 0,1,2,3,4,5,6,7,8,9,10,11,12,13,14,15);
}

// A-matrix fragment (16x32 bf16, MxK): lane m = lane&15, half = lane>>4.
// Per-lane elements: K = kk + half*16 + [0..15] (contiguous).
static __device__ __forceinline__ v16bf load_a_frag(const u16* __restrict__ base,
                                                    int row, int stride,
                                                    int kk, int half) {
  const u16* p = base + (size_t)row * stride + kk + half * 16;
  v8bf lo = *reinterpret_cast<const v8bf*>(p);
  v8bf hi = *reinterpret_cast<const v8bf*>(p + 8);
  return combine16(lo, hi);
}

static __device__ __forceinline__ v8f wmma_bf16(v16bf a, v16bf b, v8f c) {
  return __builtin_amdgcn_wmma_f32_16x16x32_bf16(false, a, false, b,
                                                 (short)0, c, false, false);
}

// ---------------------------------------------------------------------------
// Kernel 1: LN(seq_v) -> v_in (f32) + xv (bf16); seq_q/seq_k -> bf16.
// One wave per row of 512; 8 rows per 256-thread block. b128 accesses.
// ---------------------------------------------------------------------------
__global__ void __launch_bounds__(256) prep_kernel(
    const float* __restrict__ seq_q, const float* __restrict__ seq_k,
    const float* __restrict__ seq_v, const float* __restrict__ gamma,
    const float* __restrict__ beta, float* __restrict__ v_in,
    u16* __restrict__ xv, u16* __restrict__ xq, u16* __restrict__ xk) {
  const int w = threadIdx.x >> 5, lane = threadIdx.x & 31;
  const int row = blockIdx.x * 8 + w;
  const size_t base = (size_t)row * ND;

  const float4* vr4 = reinterpret_cast<const float4*>(seq_v + base);
  float x[16];
#pragma unroll
  for (int i = 0; i < 4; ++i) {
    float4 a = vr4[lane + i * 32];
    x[4 * i + 0] = a.x; x[4 * i + 1] = a.y;
    x[4 * i + 2] = a.z; x[4 * i + 3] = a.w;
  }
  float sum = 0.f;
#pragma unroll
  for (int i = 0; i < 16; ++i) sum += x[i];
#pragma unroll
  for (int off = 16; off >= 1; off >>= 1) sum += __shfl_xor(sum, off, 32);
  const float mu = sum * (1.f / (float)ND);

  float vs = 0.f;
#pragma unroll
  for (int i = 0; i < 16; ++i) { float d = x[i] - mu; vs += d * d; }
#pragma unroll
  for (int off = 16; off >= 1; off >>= 1) vs += __shfl_xor(vs, off, 32);
  const float rstd = rsqrtf(vs * (1.f / (float)ND) + 1e-5f);

  const float4* g4 = reinterpret_cast<const float4*>(gamma);
  const float4* b4 = reinterpret_cast<const float4*>(beta);
  float4* vin4 = reinterpret_cast<float4*>(v_in + base);
  ushort4* xv4 = reinterpret_cast<ushort4*>(xv + base);
#pragma unroll
  for (int i = 0; i < 4; ++i) {
    const int vi = lane + i * 32;
    float4 g = g4[vi], bb = b4[vi];
    float4 z;
    z.x = (x[4 * i + 0] - mu) * rstd * g.x + bb.x;
    z.y = (x[4 * i + 1] - mu) * rstd * g.y + bb.y;
    z.z = (x[4 * i + 2] - mu) * rstd * g.z + bb.z;
    z.w = (x[4 * i + 3] - mu) * rstd * g.w + bb.w;
    vin4[vi] = z;
    ushort4 u; u.x = f2bf(z.x); u.y = f2bf(z.y); u.z = f2bf(z.z); u.w = f2bf(z.w);
    xv4[vi] = u;
  }

  const float4* qr4 = reinterpret_cast<const float4*>(seq_q + base);
  const float4* kr4 = reinterpret_cast<const float4*>(seq_k + base);
  ushort4* xq4 = reinterpret_cast<ushort4*>(xq + base);
  ushort4* xk4 = reinterpret_cast<ushort4*>(xk + base);
#pragma unroll
  for (int i = 0; i < 4; ++i) {
    const int vi = lane + i * 32;
    float4 a = qr4[vi], c = kr4[vi];
    ushort4 uq, uk;
    uq.x = f2bf(a.x); uq.y = f2bf(a.y); uq.z = f2bf(a.z); uq.w = f2bf(a.w);
    uk.x = f2bf(c.x); uk.y = f2bf(c.y); uk.z = f2bf(c.z); uk.w = f2bf(c.w);
    xq4[vi] = uq;
    xk4[vi] = uk;
  }
}

// ---------------------------------------------------------------------------
// Kernel 2: W1/W2/W3 fp32 -> bf16 (kept row-major: a W row IS a B-column).
// ---------------------------------------------------------------------------
__global__ void __launch_bounds__(256) wconv_kernel(
    const float* __restrict__ W1, const float* __restrict__ W2,
    const float* __restrict__ W3, u16* __restrict__ wq,
    u16* __restrict__ wk, u16* __restrict__ wv) {
  const int id = blockIdx.x * 256 + threadIdx.x;   // 3 * 512*512/4 total
  const int which = id >> 16;                      // 65536 float4 per matrix
  const int rem = id & 65535;
  const float* src = (which == 0) ? W1 : (which == 1) ? W2 : W3;
  u16* dst = (which == 0) ? wq : (which == 1) ? wk : wv;
  float4 a = reinterpret_cast<const float4*>(src)[rem];
  ushort4 u; u.x = f2bf(a.x); u.y = f2bf(a.y); u.z = f2bf(a.z); u.w = f2bf(a.w);
  reinterpret_cast<ushort4*>(dst)[rem] = u;
}

// ---------------------------------------------------------------------------
// Kernel 3: q = xq@W1^T, k = xk@W2^T, v = xv@W3^T.
// grid (256, 2, 3); 8 waves = 2(M) x 4(N); wave tile 16x64.
// K-loop unrolled x2 with ping-pong buffer sets: no register rotation,
// loads for one set in flight while the other set feeds the WMMAs.
// q,k stored [b,h,s,hd] bf16; v stored transposed [b,h,hd,s] bf16.
// ---------------------------------------------------------------------------
__global__ void __launch_bounds__(256) gemm_qkv_kernel(
    const u16* __restrict__ xq, const u16* __restrict__ xk,
    const u16* __restrict__ xv, const u16* __restrict__ Wq,
    const u16* __restrict__ Wk, const u16* __restrict__ Wv,
    u16* __restrict__ qh, u16* __restrict__ kh, u16* __restrict__ vT) {
  const int which = blockIdx.z;
  const u16* x = (which == 0) ? xq : (which == 1) ? xk : xv;
  const u16* W = (which == 0) ? Wq : (which == 1) ? Wk : Wv;

  const int w = threadIdx.x >> 5, lane = threadIdx.x & 31;
  const int half = lane >> 4, l16 = lane & 15;
  const int mBase = blockIdx.x * 32 + (w & 1) * 16;
  const int nBase = blockIdx.y * 256 + (w >> 1) * 64;

  v8f acc[4] = {};
  const int rowA = mBase + l16;
  const u16* Wrow[4];
#pragma unroll
  for (int t = 0; t < 4; ++t)
    Wrow[t] = W + (size_t)(nBase + t * 16 + l16) * ND + half * 16;

  // prologue: set A <- kk = 0
  v16bf aA = load_a_frag(x, rowA, ND, 0, half);
  v16bf bA[4];
#pragma unroll
  for (int t = 0; t < 4; ++t)
    bA[t] = *reinterpret_cast<const v16bf*>(Wrow[t]);

  for (int kk = 0; kk < ND; kk += 64) {
    // set B <- kk + 32 (in flight while set A computes)
    v16bf aB = load_a_frag(x, rowA, ND, kk + 32, half);
    v16bf bB[4];
#pragma unroll
    for (int t = 0; t < 4; ++t)
      bB[t] = *reinterpret_cast<const v16bf*>(Wrow[t] + kk + 32);
#pragma unroll
    for (int t = 0; t < 4; ++t) acc[t] = wmma_bf16(aA, bA[t], acc[t]);

    // set A <- kk + 64 (wraps to 0 on last iter; result unused)
    const int kn = (kk + 64) & (ND - 1);
    aA = load_a_frag(x, rowA, ND, kn, half);
#pragma unroll
    for (int t = 0; t < 4; ++t)
      bA[t] = *reinterpret_cast<const v16bf*>(Wrow[t] + kn);
#pragma unroll
    for (int t = 0; t < 4; ++t) acc[t] = wmma_bf16(aB, bB[t], acc[t]);
  }

#pragma unroll
  for (int t = 0; t < 4; ++t) {
    const int n = nBase + t * 16 + l16;
    const int h = n >> 6, d = n & 63;
#pragma unroll
    for (int r = 0; r < 8; ++r) {
      const int R = mBase + r + half * 8;     // flattened b*S + s
      const int bb = R >> 11, s = R & 2047;
      const u16 val = f2bf(acc[t][r]);
      if (which == 2)
        vT[(size_t)(((bb * NH + h) << 6) + d) * NS + s] = val;
      else if (which == 0)
        qh[((size_t)(bb * NH + h) * NS + s) * HD + d] = val;
      else
        kh[((size_t)(bb * NH + h) * NS + s) * HD + d] = val;
    }
  }
}

// ---------------------------------------------------------------------------
// Kernel 4: flash attention. scores = K@Q^T (unscaled quirk), softmax over j,
// out = P@V. grid (S/128, H, B); each wave owns a 16(i) x 64(d) output tile.
// Q fragments issued first (drained by score WMMAs), V fragments issued
// second (drained after the softmax VALU block), next chunk prefetched.
// ---------------------------------------------------------------------------
__global__ void __launch_bounds__(256) attn_kernel(
    const u16* __restrict__ qh, const u16* __restrict__ kh,
    const u16* __restrict__ vT, float* __restrict__ att) {
  __shared__ __align__(16) u16 sP[8 * 16 * 64];   // 16KB: per-wave P tile

  const int b = blockIdx.z, h = blockIdx.y;
  const int w = threadIdx.x >> 5, lane = threadIdx.x & 31;
  const int half = lane >> 4, l16 = lane & 15;
  const int i0 = blockIdx.x * 128 + w * 16;

  const u16* qb = qh + (size_t)(b * NH + h) * NS * HD;
  const u16* kb = kh + (size_t)(b * NH + h) * NS * HD;
  const u16* vb = vT + (size_t)(b * NH + h) * HD * NS;
  u16* myP = sP + w * (16 * 64);

  // K-rows A-fragments are loop-invariant for this i-tile.
  const v16bf aK0 = load_a_frag(kb, i0 + l16, HD, 0, half);
  const v16bf aK1 = load_a_frag(kb, i0 + l16, HD, 32, half);

  v8f o[4] = {};
  float mr[8], lr[8];
#pragma unroll
  for (int r = 0; r < 8; ++r) { mr[r] = -3.0e38f; lr[r] = 0.f; }

  for (int j0 = 0; j0 < NS; j0 += 64) {
    // --- issue ALL fragment loads for this chunk up front ---------------
    // Q^T B-fragments (needed first, by the score WMMAs)
    v16bf bq[4][2];
#pragma unroll
    for (int t = 0; t < 4; ++t) {
      const u16* qrow = qb + (size_t)(j0 + t * 16 + l16) * HD + half * 16;
      bq[t][0] = *reinterpret_cast<const v16bf*>(qrow);       // d = 0..31
      bq[t][1] = *reinterpret_cast<const v16bf*>(qrow + 32);  // d = 32..63
    }
    // V B-fragments (needed last, after softmax + LDS bounce)
    v16bf bv[4][2];
#pragma unroll
    for (int t = 0; t < 4; ++t) {
      const u16* vrow = vb + (size_t)(t * 16 + l16) * NS + j0 + half * 16;
      bv[t][0] = *reinterpret_cast<const v16bf*>(vrow);       // j offs 0..31
      bv[t][1] = *reinterpret_cast<const v16bf*>(vrow + 32);  // j offs 32..63
    }
    // warm WGP$/L2 for the next chunk (global_prefetch_b8); wraps in-bounds
    {
      const int jp = (j0 + 64) & (NS - 1);
      __builtin_prefetch(qb + (size_t)(jp + lane) * HD, 0, 3);
      __builtin_prefetch(qb + (size_t)(jp + 32 + lane) * HD, 0, 3);
      __builtin_prefetch(vb + (size_t)lane * NS + jp, 0, 3);
      __builtin_prefetch(vb + (size_t)(lane + 32) * NS + jp, 0, 3);
    }

    // --- scores tile: 16(i) x 64(j), K-dim = hd = 64 (2 WMMA steps) ------
    v8f sc[4] = {};
#pragma unroll
    for (int t = 0; t < 4; ++t) {
      sc[t] = wmma_bf16(aK0, bq[t][0], sc[t]);
      sc[t] = wmma_bf16(aK1, bq[t][1], sc[t]);
    }

    // --- online softmax (row stats replicated across each 16-lane half) --
#pragma unroll
    for (int r = 0; r < 8; ++r) {
      float mx = fmaxf(fmaxf(sc[0][r], sc[1][r]), fmaxf(sc[2][r], sc[3][r]));
      mx = fmaxf(mx, __shfl_xor(mx, 1, 16));
      mx = fmaxf(mx, __shfl_xor(mx, 2, 16));
      mx = fmaxf(mx, __shfl_xor(mx, 4, 16));
      mx = fmaxf(mx, __shfl_xor(mx, 8, 16));
      const float mnew = fmaxf(mr[r], mx);
      const float p0 = __expf(sc[0][r] - mnew);
      const float p1 = __expf(sc[1][r] - mnew);
      const float p2 = __expf(sc[2][r] - mnew);
      const float p3 = __expf(sc[3][r] - mnew);
      sc[0][r] = p0; sc[1][r] = p1; sc[2][r] = p2; sc[3][r] = p3;
      float rs = (p0 + p1) + (p2 + p3);
      rs += __shfl_xor(rs, 1, 16);
      rs += __shfl_xor(rs, 2, 16);
      rs += __shfl_xor(rs, 4, 16);
      rs += __shfl_xor(rs, 8, 16);
      const float scale = __expf(mr[r] - mnew);
      lr[r] = lr[r] * scale + rs;
      mr[r] = mnew;
      o[0][r] *= scale; o[1][r] *= scale; o[2][r] *= scale; o[3][r] *= scale;
    }

    // --- bounce P through LDS: C-layout -> A-layout (in-order per wave) --
#pragma unroll
    for (int t = 0; t < 4; ++t)
#pragma unroll
      for (int r = 0; r < 8; ++r)
        myP[(r + half * 8) * 64 + t * 16 + l16] = f2bf(sc[t][r]);

    v16bf aP0, aP1;
    {
      const u16* p = myP + l16 * 64;
      aP0 = combine16(*reinterpret_cast<const v8bf*>(p + 0 + half * 16),
                      *reinterpret_cast<const v8bf*>(p + 8 + half * 16));
      aP1 = combine16(*reinterpret_cast<const v8bf*>(p + 32 + half * 16),
                      *reinterpret_cast<const v8bf*>(p + 40 + half * 16));
    }

    // --- o += P @ V (V transposed: B[k=j][n=d] = vT[d*S + j]) ------------
#pragma unroll
    for (int t = 0; t < 4; ++t) {
      o[t] = wmma_bf16(aP0, bv[t][0], o[t]);
      o[t] = wmma_bf16(aP1, bv[t][1], o[t]);
    }
  }

  // epilogue: normalize by softmax denom, scatter heads back to [b,s,D]
#pragma unroll
  for (int t = 0; t < 4; ++t) {
    const int dg = h * HD + t * 16 + l16;
#pragma unroll
    for (int r = 0; r < 8; ++r) {
      const int s = i0 + r + half * 8;
      att[((size_t)b * NS + s) * ND + dg] = o[t][r] / lr[r];
    }
  }
}

// ---------------------------------------------------------------------------
// Kernel 5: out = LN(att + v_in) -> d_out (fp32)
// ---------------------------------------------------------------------------
__global__ void __launch_bounds__(256) final_kernel(
    const float* __restrict__ att, const float* __restrict__ v_in,
    const float* __restrict__ gamma, const float* __restrict__ beta,
    float* __restrict__ out) {
  const int w = threadIdx.x >> 5, lane = threadIdx.x & 31;
  const int row = blockIdx.x * 8 + w;
  const size_t base = (size_t)row * ND;

  const float4* a4 = reinterpret_cast<const float4*>(att + base);
  const float4* v4 = reinterpret_cast<const float4*>(v_in + base);
  float x[16];
#pragma unroll
  for (int i = 0; i < 4; ++i) {
    const int vi = lane + i * 32;
    float4 a = a4[vi], v = v4[vi];
    x[4 * i + 0] = a.x + v.x; x[4 * i + 1] = a.y + v.y;
    x[4 * i + 2] = a.z + v.z; x[4 * i + 3] = a.w + v.w;
  }
  float sum = 0.f;
#pragma unroll
  for (int i = 0; i < 16; ++i) sum += x[i];
#pragma unroll
  for (int off = 16; off >= 1; off >>= 1) sum += __shfl_xor(sum, off, 32);
  const float mu = sum * (1.f / (float)ND);

  float vs = 0.f;
#pragma unroll
  for (int i = 0; i < 16; ++i) { float d = x[i] - mu; vs += d * d; }
#pragma unroll
  for (int off = 16; off >= 1; off >>= 1) vs += __shfl_xor(vs, off, 32);
  const float rstd = rsqrtf(vs * (1.f / (float)ND) + 1e-5f);

  const float4* g4 = reinterpret_cast<const float4*>(gamma);
  const float4* b4 = reinterpret_cast<const float4*>(beta);
  float4* o4 = reinterpret_cast<float4*>(out + base);
#pragma unroll
  for (int i = 0; i < 4; ++i) {
    const int vi = lane + i * 32;
    float4 g = g4[vi], bb = b4[vi];
    float4 z;
    z.x = (x[4 * i + 0] - mu) * rstd * g.x + bb.x;
    z.y = (x[4 * i + 1] - mu) * rstd * g.y + bb.y;
    z.z = (x[4 * i + 2] - mu) * rstd * g.z + bb.z;
    z.w = (x[4 * i + 3] - mu) * rstd * g.w + bb.w;
    o4[vi] = z;
  }
}

// ---------------------------------------------------------------------------
// Host-side launcher.  Workspace layout (bytes), ~65.5 MB total; `att`
// aliases the dead xv/xq buffers (projections consume them before attn runs).
// ---------------------------------------------------------------------------
extern "C" void kernel_launch(void* const* d_in, const int* in_sizes, int n_in,
                              void* d_out, int out_size, void* d_ws,
                              size_t ws_size, hipStream_t stream) {
  const float* seq_k = (const float*)d_in[0];
  const float* seq_q = (const float*)d_in[1];
  const float* seq_v = (const float*)d_in[2];
  const float* W1    = (const float*)d_in[3];
  const float* W2    = (const float*)d_in[4];
  const float* W3    = (const float*)d_in[5];
  const float* gamma = (const float*)d_in[6];
  const float* beta  = (const float*)d_in[7];
  float* out = (float*)d_out;

  char* ws = (char*)d_ws;
  const size_t NTOK = (size_t)NB * NS;            // 8192 rows
  const size_t F32  = NTOK * ND * 4;              // 16 MB
  const size_t BF16 = NTOK * ND * 2;              // 8 MB
  const size_t WSZ  = (size_t)ND * ND * 2;        // 0.5 MB

  float* v_in = (float*)(ws);                          // [B*S, D] f32
  u16* xv = (u16*)(ws + F32);                          // LN(v) bf16
  u16* xq = (u16*)(ws + F32 + BF16);                   // q input bf16
  u16* xk = (u16*)(ws + F32 + 2 * BF16);               // k input bf16
  u16* wq = (u16*)(ws + F32 + 3 * BF16);               // W1 bf16
  u16* wk = (u16*)(ws + F32 + 3 * BF16 + WSZ);         // W2 bf16
  u16* wv = (u16*)(ws + F32 + 3 * BF16 + 2 * WSZ);     // W3 bf16
  u16* qh = (u16*)(ws + F32 + 3 * BF16 + 3 * WSZ);     // [b,h,s,hd]
  u16* kh = (u16*)(ws + F32 + 4 * BF16 + 3 * WSZ);     // [b,h,s,hd]
  u16* vT = (u16*)(ws + F32 + 5 * BF16 + 3 * WSZ);     // [b,h,hd,s]
  float* att = (float*)(ws + F32);                     // aliases xv+xq (16MB)

  prep_kernel<<<NTOK / 8, 256, 0, stream>>>(seq_q, seq_k, seq_v, gamma, beta,
                                            v_in, xv, xq, xk);
  wconv_kernel<<<(3 * ND * ND / 4) / 256, 256, 0, stream>>>(W1, W2, W3,
                                                            wq, wk, wv);
  gemm_qkv_kernel<<<dim3(NTOK / 32, ND / 256, 3), 256, 0, stream>>>(
      xq, xk, xv, wq, wk, wv, qh, kh, vT);
  attn_kernel<<<dim3(NS / 128, NH, NB), 256, 0, stream>>>(qh, kh, vT, att);
  final_kernel<<<NTOK / 8, 256, 0, stream>>>(att, v_in, gamma, beta, out);
}
